// STGNN_12532714570115
// MI455X (gfx1250) — compile-verified, matrix operand order checked by reading, not run on hardware
//
#include <hip/hip_runtime.h>

typedef __attribute__((ext_vector_type(16))) _Float16 v16h;
typedef __attribute__((ext_vector_type(8)))  float    v8f;
typedef unsigned int u32;

#define NN    20000
#define TT    256
#define EE    320000
#define GG    64
#define CTC   8
#define CPC   16
#define EMB   768
#define FINF  512
#define OUTC  4

// ---------------- hardware f32 atomic add (non-returning) ----------------
__device__ __forceinline__ void atomic_add_f32(float* addr, float val) {
  asm volatile("global_atomic_add_f32 %0, %1, off" : : "v"(addr), "v"(val) : "memory");
}

// ---------------- zero ----------------
__global__ void zero_kernel(float* __restrict__ p, long n) {
  long i = (long)blockIdx.x * blockDim.x + threadIdx.x;
  long stride = (long)gridDim.x * blockDim.x;
  for (; i < n; i += stride) p[i] = 0.0f;
}

// ---------------- weight transpose + f16 convert: w(K x Nc) -> wt(Nc x K) ----------------
__global__ void transpose_f16(const float* __restrict__ w, _Float16* __restrict__ wt,
                              int K, int Nc) {
  long idx = (long)blockIdx.x * blockDim.x + threadIdx.x;
  if (idx >= (long)K * Nc) return;
  int n = (int)(idx % Nc);
  int k = (int)(idx / Nc);
  wt[(long)n * K + k] = (_Float16)w[idx];
}

// ---------------- fused temporal CNN per node ----------------
__global__ __launch_bounds__(256) void temporal_kernel(
    const float* __restrict__ x,
    const float* __restrict__ w0,
    const float* __restrict__ bn0g, const float* __restrict__ bn0b,
    const float* __restrict__ bn0m, const float* __restrict__ bn0v,
    const float* __restrict__ w1,
    const float* __restrict__ w2,
    const float* __restrict__ bn2g, const float* __restrict__ bn2b,
    const float* __restrict__ bn2m, const float* __restrict__ bn2v,
    float* __restrict__ h0) {
  __shared__ float xs[TT + 32];
  __shared__ float w0s[33 * CTC];
  __shared__ float w1s[21 * CTC];
  __shared__ float w2s[CTC * CPC];
  __shared__ float sc0[CTC], sb0[CTC], sc2[CPC], sb2[CPC];
  __shared__ float s1[TT + 20][CTC + 1];
  __shared__ float es[TT][CPC + 1];

  const int n = blockIdx.x;
  const int t = threadIdx.x;

  xs[16 + t] = x[(long)n * TT + t];
  if (t < 16) { xs[t] = 0.0f; xs[TT + 16 + t] = 0.0f; }
  if (t < 33 * CTC) w0s[t] = w0[t];
  if (t < 21 * CTC) w1s[t] = w1[t];
  if (t < CTC * CPC) w2s[t] = w2[t];
  if (t < CTC) {
    float s = bn0g[t] * rsqrtf(bn0v[t] + 1e-5f);
    sc0[t] = s; sb0[t] = bn0b[t] - bn0m[t] * s;
  }
  if (t < CPC) {
    float s = bn2g[t] * rsqrtf(bn2v[t] + 1e-5f);
    sc2[t] = s; sb2[t] = bn2b[t] - bn2m[t] * s;
  }
  if (t < 10) {
    #pragma unroll
    for (int c = 0; c < CTC; ++c) { s1[t][c] = 0.0f; s1[TT + 10 + t][c] = 0.0f; }
  }
  __syncthreads();

  // conv0 (k=33, in=1, out=8) + bn0
  float acc[CTC];
  #pragma unroll
  for (int c = 0; c < CTC; ++c) acc[c] = 0.0f;
  for (int k = 0; k < 33; ++k) {
    float xv = xs[t + k];
    #pragma unroll
    for (int c = 0; c < CTC; ++c) acc[c] += xv * w0s[k * CTC + c];
  }
  #pragma unroll
  for (int c = 0; c < CTC; ++c) s1[10 + t][c] = acc[c] * sc0[c] + sb0[c];
  __syncthreads();

  // depthwise conv1 (k=21) + relu
  float d[CTC];
  #pragma unroll
  for (int c = 0; c < CTC; ++c) d[c] = 0.0f;
  for (int k = 0; k < 21; ++k) {
    #pragma unroll
    for (int c = 0; c < CTC; ++c) d[c] += s1[t + k][c] * w1s[k * CTC + c];
  }
  #pragma unroll
  for (int c = 0; c < CTC; ++c) d[c] = fmaxf(d[c], 0.0f);

  // 1x1 conv2 (8 -> 16) + bn2 + relu
  float e[CPC];
  #pragma unroll
  for (int p = 0; p < CPC; ++p) e[p] = 0.0f;
  #pragma unroll
  for (int c = 0; c < CTC; ++c) {
    float dv = d[c];
    #pragma unroll
    for (int p = 0; p < CPC; ++p) e[p] += dv * w2s[c * CPC + p];
  }
  #pragma unroll
  for (int p = 0; p < CPC; ++p) es[t][p] = fmaxf(e[p] * sc2[p] + sb2[p], 0.0f);
  __syncthreads();

  // avg-pool over 8 time steps -> (32 blocks x 16 ch) = 512 features
  for (int o = t; o < FINF; o += 256) {
    int tb = o >> 4, p = o & 15;
    float s = 0.0f;
    #pragma unroll
    for (int j = 0; j < 8; ++j) s += es[tb * 8 + j][p];
    h0[(long)n * FINF + o] = s * 0.125f;
  }
}

// ---------------- edge scatter-add: agg[dst] += h[src] ----------------
template <int F>
__global__ void edge_scatter(const int* __restrict__ ei,
                             const float* __restrict__ h,
                             float* __restrict__ agg) {
  long idx = (long)blockIdx.x * blockDim.x + threadIdx.x;
  const int q = F >> 2;
  if (idx >= (long)EE * q) return;
  int e = (int)(idx / q);
  int f = (int)(idx % q) << 2;
  int s = ei[e];
  int d = ei[EE + e];
  const float4 v = *(const float4*)(h + (long)s * F + f);
  float* dp = agg + (long)d * F + f;
  atomic_add_f32(dp + 0, v.x);
  atomic_add_f32(dp + 1, v.y);
  atomic_add_f32(dp + 2, v.z);
  atomic_add_f32(dp + 3, v.w);
}

// ---------------- u = h + agg, converted to f16 ----------------
__global__ void add_convert(const float* __restrict__ h, const float* __restrict__ agg,
                            _Float16* __restrict__ out, long n) {
  long i = (long)blockIdx.x * blockDim.x + threadIdx.x;
  long stride = (long)gridDim.x * blockDim.x;
  for (; i < n; i += stride) out[i] = (_Float16)(h[i] + agg[i]);
}

// ---------------- WMMA fragment load (ISA 16-bit 16x32 A/B layout) ----------------
// VGPRs 0-3: K = base+0..7 ; VGPRs 4-7: K = base+16..23 (pairs of f16 per dword)
__device__ __forceinline__ v16h load_frag(const u32* __restrict__ p) {
  union { u32 u[8]; v16h h; } t;
  t.u[0] = p[0];  t.u[1] = p[1];  t.u[2] = p[2];  t.u[3] = p[3];
  t.u[4] = p[8];  t.u[5] = p[9];  t.u[6] = p[10]; t.u[7] = p[11];
  return t.h;
}

// ---------------- WMMA GEMM: C = relu(A(MxK) * Wt(NcxK)^T + bias) ----------------
// One wave per 16x64 output strip: one A-fragment reused across 4 B-fragments
// (4 independent accumulators -> no WMMA->WMMA RAW hazard, 2.5 loads per WMMA).
__global__ __launch_bounds__(256) void wmma_gemm_relu(
    const _Float16* __restrict__ A, const _Float16* __restrict__ Wt,
    const float* __restrict__ bias,
    _Float16* __restrict__ outH, float* __restrict__ outF,
    int M, int Nc, int K) {
  int wave = (int)((blockIdx.x * blockDim.x + threadIdx.x) >> 5);
  int lane = threadIdx.x & 31;
  int n64 = Nc >> 6;                 // 64-wide column strips
  int tM = wave / n64;
  int tN = wave - tM * n64;
  if (tM * 16 >= M) return;  // uniform per wave: EXEC stays all-ones for WMMA

  int rowA = tM * 16 + (lane & 15);
  int colL = (lane & 15);
  int pbase = (lane >> 4) * 4;       // half-wave selects K-base 0 / 8 (f16 pairs)
  const u32* Ap  = (const u32*)(A  + (long)rowA * K) + pbase;
  const u32* Bp0 = (const u32*)(Wt + (long)(tN * 64 +  0 + colL) * K) + pbase;
  const u32* Bp1 = (const u32*)(Wt + (long)(tN * 64 + 16 + colL) * K) + pbase;
  const u32* Bp2 = (const u32*)(Wt + (long)(tN * 64 + 32 + colL) * K) + pbase;
  const u32* Bp3 = (const u32*)(Wt + (long)(tN * 64 + 48 + colL) * K) + pbase;

  v8f acc0 = {}, acc1 = {}, acc2 = {}, acc3 = {};
  int steps = K >> 5;                // 32 K-elements per WMMA
  for (int s = 0; s < steps; ++s) {
    v16h a  = load_frag(Ap  + s * 16);
    v16h b0 = load_frag(Bp0 + s * 16);
    v16h b1 = load_frag(Bp1 + s * 16);
    v16h b2 = load_frag(Bp2 + s * 16);
    v16h b3 = load_frag(Bp3 + s * 16);
    acc0 = __builtin_amdgcn_wmma_f32_16x16x32_f16(false, a, false, b0, (short)0, acc0, false, false);
    acc1 = __builtin_amdgcn_wmma_f32_16x16x32_f16(false, a, false, b1, (short)0, acc1, false, false);
    acc2 = __builtin_amdgcn_wmma_f32_16x16x32_f16(false, a, false, b2, (short)0, acc2, false, false);
    acc3 = __builtin_amdgcn_wmma_f32_16x16x32_f16(false, a, false, b3, (short)0, acc3, false, false);
  }

  int rbase = tM * 16 + ((lane >> 4) << 3);
  #pragma unroll
  for (int j = 0; j < 4; ++j) {
    v8f acc = (j == 0) ? acc0 : (j == 1) ? acc1 : (j == 2) ? acc2 : acc3;
    int col = tN * 64 + j * 16 + colL;
    float bv = bias[col];
    #pragma unroll
    for (int i = 0; i < 8; ++i) {
      float r = fmaxf(acc[i] + bv, 0.0f);
      long off = (long)(rbase + i) * Nc + col;
      if (outF) outF[off] = r;
      if (outH) outH[off] = (_Float16)r;
    }
  }
}

// ---------------- graph pooling ----------------
__global__ void pool_scatter(const float* __restrict__ h, const int* __restrict__ batch,
                             float* __restrict__ pooled) {
  long idx = (long)blockIdx.x * blockDim.x + threadIdx.x;
  const int q = EMB >> 2;
  if (idx >= (long)NN * q) return;
  int n = (int)(idx / q);
  int f = (int)(idx % q) << 2;
  int b = batch[n];
  const float4 v = *(const float4*)(h + (long)n * EMB + f);
  float* dp = pooled + (long)b * EMB + f;
  atomic_add_f32(dp + 0, v.x);
  atomic_add_f32(dp + 1, v.y);
  atomic_add_f32(dp + 2, v.z);
  atomic_add_f32(dp + 3, v.w);
}

__global__ void count_kernel(const int* __restrict__ batch, float* __restrict__ cnt) {
  int n = blockIdx.x * blockDim.x + threadIdx.x;
  if (n < NN) atomic_add_f32(&cnt[batch[n]], 1.0f);
}

// ---------------- dense + mean + log_softmax ----------------
__global__ __launch_bounds__(128) void final_kernel(
    const float* __restrict__ pooled, const float* __restrict__ cnt,
    const float* __restrict__ dw, const float* __restrict__ db,
    float* __restrict__ out) {
  __shared__ float red[128][OUTC];
  int g = blockIdx.x, t = threadIdx.x;
  float a[OUTC] = {0.0f, 0.0f, 0.0f, 0.0f};
  for (int k = t; k < EMB; k += 128) {
    float p = pooled[(long)g * EMB + k];
    #pragma unroll
    for (int o = 0; o < OUTC; ++o) a[o] += p * dw[k * OUTC + o];
  }
  #pragma unroll
  for (int o = 0; o < OUTC; ++o) red[t][o] = a[o];
  __syncthreads();
  for (int s = 64; s > 0; s >>= 1) {
    if (t < s) {
      #pragma unroll
      for (int o = 0; o < OUTC; ++o) red[t][o] += red[t + s][o];
    }
    __syncthreads();
  }
  if (t == 0) {
    float c = fmaxf(cnt[g], 1.0f);
    float l[OUTC];
    float m = -3.4e38f;
    #pragma unroll
    for (int o = 0; o < OUTC; ++o) { l[o] = red[0][o] / c + db[o]; m = fmaxf(m, l[o]); }
    float s = 0.0f;
    #pragma unroll
    for (int o = 0; o < OUTC; ++o) s += expf(l[o] - m);
    float lse = logf(s);
    #pragma unroll
    for (int o = 0; o < OUTC; ++o) out[g * OUTC + o] = l[o] - m - lse;
  }
}

// ---------------- launch ----------------
static inline long cdivl(long a, long b) { return (a + b - 1) / b; }

extern "C" void kernel_launch(void* const* d_in, const int* in_sizes, int n_in,
                              void* d_out, int out_size, void* d_ws, size_t ws_size,
                              hipStream_t stream) {
  const float* x        = (const float*)d_in[0];
  const int*   ei       = (const int*)d_in[1];
  const int*   batch    = (const int*)d_in[2];
  const float* conv0_w  = (const float*)d_in[3];
  const float* bn0g     = (const float*)d_in[4];
  const float* bn0b     = (const float*)d_in[5];
  const float* bn0m     = (const float*)d_in[6];
  const float* bn0v     = (const float*)d_in[7];
  const float* conv1_w  = (const float*)d_in[8];
  const float* conv2_w  = (const float*)d_in[9];
  const float* bn2g     = (const float*)d_in[10];
  const float* bn2b     = (const float*)d_in[11];
  const float* bn2m     = (const float*)d_in[12];
  const float* bn2v     = (const float*)d_in[13];
  const float* gin1_w1  = (const float*)d_in[14];
  const float* gin1_b1  = (const float*)d_in[15];
  const float* gin1_w2  = (const float*)d_in[16];
  const float* gin1_b2  = (const float*)d_in[17];
  const float* gin2_w1  = (const float*)d_in[18];
  const float* gin2_b1  = (const float*)d_in[19];
  const float* gin2_w2  = (const float*)d_in[20];
  const float* gin2_b2  = (const float*)d_in[21];
  const float* dense_w  = (const float*)d_in[22];
  const float* dense_b  = (const float*)d_in[23];
  float* out = (float*)d_out;

  // workspace layout (256B-aligned slices)
  char* ws = (char*)d_ws;
  size_t off = 0;
  auto alloc = [&](size_t bytes) { size_t o = off; off += (bytes + 255) & ~(size_t)255; return o; };
  float*    h0     = (float*)(ws + alloc((size_t)NN * FINF * 4));
  float*    h1     = (float*)(ws + alloc((size_t)NN * EMB * 4));
  float*    agg    = (float*)(ws + alloc((size_t)NN * EMB * 4));  // also reused as h2
  _Float16* bufA   = (_Float16*)(ws + alloc((size_t)NN * EMB * 2));
  _Float16* bufB   = (_Float16*)(ws + alloc((size_t)NN * EMB * 2));
  _Float16* w1t    = (_Float16*)(ws + alloc((size_t)EMB * FINF * 2));
  _Float16* w2t    = (_Float16*)(ws + alloc((size_t)EMB * EMB * 2));
  _Float16* w3t    = (_Float16*)(ws + alloc((size_t)EMB * EMB * 2));
  _Float16* w4t    = (_Float16*)(ws + alloc((size_t)EMB * EMB * 2));
  float*    pooled = (float*)(ws + alloc((size_t)GG * EMB * 4));
  float*    cnt    = (float*)(ws + alloc((size_t)GG * 4));

  // 1. weight transposes (f32 -> f16, K-contiguous per output column)
  transpose_f16<<<cdivl((long)FINF * EMB, 256), 256, 0, stream>>>(gin1_w1, w1t, FINF, EMB);
  transpose_f16<<<cdivl((long)EMB * EMB, 256), 256, 0, stream>>>(gin1_w2, w2t, EMB, EMB);
  transpose_f16<<<cdivl((long)EMB * EMB, 256), 256, 0, stream>>>(gin2_w1, w3t, EMB, EMB);
  transpose_f16<<<cdivl((long)EMB * EMB, 256), 256, 0, stream>>>(gin2_w2, w4t, EMB, EMB);

  // 2. fused temporal CNN -> h0 (N x 512)
  temporal_kernel<<<NN, 256, 0, stream>>>(x, conv0_w, bn0g, bn0b, bn0m, bn0v,
                                          conv1_w, conv2_w, bn2g, bn2b, bn2m, bn2v, h0);

  // 3-6. GIN layer 1
  zero_kernel<<<2048, 256, 0, stream>>>(agg, (long)NN * FINF);
  edge_scatter<FINF><<<cdivl((long)EE * (FINF / 4), 256), 256, 0, stream>>>(ei, h0, agg);
  add_convert<<<cdivl((long)NN * FINF, 256), 256, 0, stream>>>(h0, agg, bufA, (long)NN * FINF);
  {
    long waves = ((long)NN / 16) * (EMB / 64);
    wmma_gemm_relu<<<cdivl(waves, 8), 256, 0, stream>>>(bufA, w1t, gin1_b1, bufB, nullptr,
                                                        NN, EMB, FINF);
    wmma_gemm_relu<<<cdivl(waves, 8), 256, 0, stream>>>(bufB, w2t, gin1_b2, nullptr, h1,
                                                        NN, EMB, EMB);
  }

  // 7-10. GIN layer 2 (h2 written into agg buffer)
  zero_kernel<<<2048, 256, 0, stream>>>(agg, (long)NN * EMB);
  edge_scatter<EMB><<<cdivl((long)EE * (EMB / 4), 256), 256, 0, stream>>>(ei, h1, agg);
  add_convert<<<cdivl((long)NN * EMB, 256), 256, 0, stream>>>(h1, agg, bufA, (long)NN * EMB);
  {
    long waves = ((long)NN / 16) * (EMB / 64);
    wmma_gemm_relu<<<cdivl(waves, 8), 256, 0, stream>>>(bufA, w3t, gin2_b1, bufB, nullptr,
                                                        NN, EMB, EMB);
    wmma_gemm_relu<<<cdivl(waves, 8), 256, 0, stream>>>(bufB, w4t, gin2_b2, nullptr, agg,
                                                        NN, EMB, EMB);
  }

  // 11. graph mean pool + dense + log_softmax
  zero_kernel<<<256, 256, 0, stream>>>(pooled, (long)GG * EMB);
  zero_kernel<<<1, 256, 0, stream>>>(cnt, (long)GG);
  pool_scatter<<<cdivl((long)NN * (EMB / 4), 256), 256, 0, stream>>>(agg, batch, pooled);
  count_kernel<<<cdivl(NN, 256), 256, 0, stream>>>(batch, cnt);
  final_kernel<<<GG, 128, 0, stream>>>(pooled, cnt, dense_w, dense_b, out);
}
